// N2PAttention_22436909154640
// MI455X (gfx1250) — matrix operand order repaired and checked by reading, not compiled, not run on hardware
//
#include <hip/hip_runtime.h>
#include <hip/hip_bf16.h>
#include <stdint.h>

#define B_ 8
#define C_ 128
#define N_ 2048
#define H_ 4
#define K_ 32
#define F_ 512
#define EPS_ 1e-5f

typedef __bf16 bf16_t;
typedef bf16_t v16bf __attribute__((ext_vector_type(16)));
typedef float  v8f   __attribute__((ext_vector_type(8)));

#define WMMA_BF16(a, b, c) \
  __builtin_amdgcn_wmma_f32_16x16x32_bf16(false, (a), false, (b), (short)0, (c), false, false)

// A-matrix 16x32 bf16 tile, row-major source with leading dim ld.
// Lane L<16: M=L, K in {0..7, 16..23}; lane L>=16: M=L-16, K in {8..15, 24..31}.
static __device__ inline v16bf load_A16x32(const bf16_t* __restrict__ src, int ld) {
  int lane = threadIdx.x & 31;
  int m = lane & 15, hi = lane >> 4;
  const bf16_t* r = src + (size_t)m * ld + hi * 8;
  v16bf a;
#pragma unroll
  for (int i = 0; i < 8; i++) { a[i] = r[i]; a[i + 8] = r[i + 16]; }
  return a;
}

// B-matrix 32x16 bf16 tile where element (k, n) lives at src[n*ld + k].
// Lanes 0..15 hold K in {0..7, 16..23} of column n=lane; lanes 16..31 hold K in {8..15, 24..31} of n=lane-16.
static __device__ inline v16bf load_B32x16(const bf16_t* __restrict__ src, int ld) {
  int lane = threadIdx.x & 31;
  int n = lane & 15, hi = lane >> 4;
  const bf16_t* r = src + (size_t)n * ld + hi * 8;
  v16bf b;
#pragma unroll
  for (int i = 0; i < 8; i++) { b[i] = r[i]; b[i + 8] = r[i + 16]; }
  return b;
}

// C/D 16x16 f32 tile coords: VGPR r -> row mb+r, lane -> column n.
static __device__ inline void cd_coords(int& mb, int& n) {
  int lane = threadIdx.x & 31;
  n = lane & 15;
  mb = (lane >> 4) * 8;
}

// float -> order-preserving u32 key (ascending), low 11 bits free for an index
static __device__ inline unsigned sortable_key(float f, unsigned idx11) {
  unsigned u = __float_as_uint(f);
  unsigned key = u ^ (unsigned)(((int)u >> 31) | 0x80000000);
  return (key & 0xFFFFF800u) | idx11;
}

static __device__ inline unsigned umin2(unsigned a, unsigned b) { return a < b ? a : b; }

// ---------------- prep ----------------

// tiled transpose: x [B,C,N] -> xt/xt_bf [B,N,C], all accesses coalesced
__global__ void k_transpose(const float* __restrict__ x, float* __restrict__ xt,
                            bf16_t* __restrict__ xt_bf) {
  __shared__ float tile[32][33];
  int n0 = blockIdx.x * 32, c0 = blockIdx.y * 32, b = blockIdx.z;
  int tx = threadIdx.x & 31, ty = threadIdx.x >> 5;   // 8 rows per pass
#pragma unroll
  for (int r = 0; r < 32; r += 8)
    tile[ty + r][tx] = x[((size_t)(b * C_ + c0 + ty + r)) * N_ + n0 + tx];
  __syncthreads();
#pragma unroll
  for (int r = 0; r < 32; r += 8) {
    float v = tile[tx][ty + r];                       // [c_local][n_local]
    size_t o = ((size_t)(b * N_ + n0 + ty + r)) * C_ + c0 + tx;
    xt[o] = v;
    xt_bf[o] = (bf16_t)v;
  }
}

__global__ void k_x2(const float* __restrict__ xt, float* __restrict__ x2) {
  int wave = threadIdx.x >> 5, lane = threadIdx.x & 31;
  int p = blockIdx.x * 8 + wave;                      // point in [0, B*N)
  const float* row = xt + (size_t)p * C_;
  float s = 0.f;
  for (int c = lane; c < C_; c += 32) { float v = row[c]; s += v * v; }
  for (int off = 16; off; off >>= 1) s += __shfl_xor(s, off, 32);
  if (lane == 0) x2[p] = s;
}

__global__ void k_f2bf(const float* __restrict__ src, bf16_t* __restrict__ dst, int n) {
  int i = blockIdx.x * 256 + threadIdx.x;
  if (i < n) dst[i] = (bf16_t)src[i];
}

// ---------------- KNN: Gram -> sortable score keys + top-K ----------------

// key[b][i][j] encodes score = x2[j] - 2*<xt_i, xt_j> with j in the low 11 bits
__global__ void k_gram(const bf16_t* __restrict__ xt_bf, const float* __restrict__ x2,
                       unsigned* __restrict__ S) {
  int b = blockIdx.y;
  int i0 = blockIdx.x * 16;
  int wave = threadIdx.x >> 5;
  const bf16_t* xtb = xt_bf + (size_t)b * N_ * C_;
  v16bf a[4];
#pragma unroll
  for (int kt = 0; kt < 4; kt++)
    a[kt] = load_A16x32(xtb + (size_t)i0 * C_ + kt * 32, C_);
  int mb, nl; cd_coords(mb, nl);
  for (int jt = wave; jt < N_ / 16; jt += 8) {
    int j0 = jt * 16;
    v8f acc = {};
#pragma unroll
    for (int kt = 0; kt < 4; kt++) {
      v16bf bm = load_B32x16(xtb + (size_t)j0 * C_ + kt * 32, C_);
      acc = WMMA_BF16(a[kt], bm, acc);
    }
    int j = j0 + nl;
    float xj = x2[b * N_ + j];
    unsigned* Sr = S + ((size_t)(b * N_ + i0)) * N_;
#pragma unroll
    for (int r = 0; r < 8; r++)
      Sr[(size_t)(mb + r) * N_ + j] = sortable_key(xj - 2.0f * acc[r], (unsigned)j);
  }
}

// one wave per row; per-chunk(32) minima cached in regs; only the winning chunk is rescanned
__global__ void k_topk(const unsigned* __restrict__ S, int* __restrict__ idx) {
  __shared__ unsigned Sl[4][N_];
  int wave = threadIdx.x >> 5, lane = threadIdx.x & 31;
  int row = blockIdx.x * 4 + wave;                    // in [0, B*N)
  const unsigned* Sr = S + (size_t)row * N_;
  unsigned cm0 = 0xFFFFFFFFu, cm1 = 0xFFFFFFFFu;      // chunk s owner: lane s>>1, slot s&1
#pragma unroll
  for (int s = 0; s < 64; s++) {
    unsigned key = Sr[s * 32 + lane];
    Sl[wave][s * 32 + lane] = key;
    unsigned m = key;
    for (int off = 16; off; off >>= 1)
      m = umin2(m, (unsigned)__shfl_xor((int)m, off, 32));
    if (lane == (s >> 1)) { if (s & 1) cm1 = m; else cm0 = m; }
  }
  __syncthreads();
  for (int it = 0; it < K_; it++) {
    unsigned k = umin2(cm0, cm1);
    for (int off = 16; off; off >>= 1)
      k = umin2(k, (unsigned)__shfl_xor((int)k, off, 32));
    int widx = (int)(k & 2047u);
    if (lane == 0) idx[(size_t)row * K_ + it] = widx;
    // rescan the winning chunk with the winner invalidated
    int cbase = widx & ~31;
    unsigned v = Sl[wave][cbase + lane];
    if (cbase + lane == widx) { v = 0xFFFFFFFFu; Sl[wave][widx] = 0xFFFFFFFFu; }
    unsigned m = v;
    for (int off = 16; off; off >>= 1)
      m = umin2(m, (unsigned)__shfl_xor((int)m, off, 32));
    int owner = widx >> 6, slot = (widx >> 5) & 1;
    if (lane == owner) { if (slot) cm1 = m; else cm0 = m; }
    __syncthreads();
  }
}

// ---------------- fused neighbor attention (one point per block, 4 waves) ----------------

__global__ void k_attn(const float* __restrict__ x, const float* __restrict__ xt,
                       const int* __restrict__ idx, const float* __restrict__ wq,
                       const bf16_t* __restrict__ wk_bf, const bf16_t* __restrict__ wv_bf,
                       float* __restrict__ r1) {
  __shared__ float  xi[C_];
  __shared__ bf16_t diffb[K_][C_];
  __shared__ float  kmat[K_][C_];
  __shared__ float  vmat[K_][C_];
  __shared__ float  qv[C_];
  __shared__ float  attns[H_][K_];
  __shared__ int    nidx[K_];
  int t = threadIdx.x;
  int wave = t >> 5, lane = t & 31;
  int pid = blockIdx.x;
  int b = pid >> 11;                                  // N_ == 2048
  int i = pid & (N_ - 1);
  const float* xtb = xt + (size_t)b * N_ * C_;
  xi[t] = xtb[(size_t)i * C_ + t];
  if (t < K_) nidx[t] = idx[(size_t)pid * K_ + t];
  __syncthreads();
  for (int e = t; e < K_ * C_; e += 128) {
    int kk = e >> 7, c = e & (C_ - 1);
    int j = nidx[kk];
    diffb[kk][c] = (bf16_t)(xtb[(size_t)j * C_ + c] - xi[c]);
  }
  __syncthreads();
  // K/V projections: out[m][o] = sum_c diff[m][c] * W[o][c]; 32 tile-jobs over 4 waves
  int mb, nl; cd_coords(mb, nl);
  for (int s = 0; s < 8; s++) {
    int job = wave * 8 + s;
    int proj = job >> 4, mt = (job >> 3) & 1, nt = job & 7;
    const bf16_t* W = proj ? wv_bf : wk_bf;
    v8f acc = {};
#pragma unroll
    for (int kt = 0; kt < 4; kt++) {
      v16bf a  = load_A16x32(&diffb[mt * 16][0] + kt * 32, C_);
      v16bf bm = load_B32x16(W + (size_t)(nt * 16) * C_ + kt * 32, C_);
      acc = WMMA_BF16(a, bm, acc);
    }
    float* D = proj ? &vmat[0][0] : &kmat[0][0];
#pragma unroll
    for (int r = 0; r < 8; r++)
      D[(mt * 16 + mb + r) * C_ + nt * 16 + nl] = acc[r];
  }
  __syncthreads();
  // q[o] = <xi, wq[o,:]> in f32
  {
    float s = 0.f;
    const float* wr = wq + (size_t)t * C_;
    for (int ci = 0; ci < C_; ci++) s += xi[ci] * wr[ci];
    qv[t] = s;
  }
  __syncthreads();
  // energy + softmax: head h = wave, neighbor kk = lane
  {
    int h = wave, kk = lane;
    float e = 0.f;
    for (int d = 0; d < 32; d++) e += qv[h * 32 + d] * kmat[kk][h * 32 + d];
    e *= 0.1767766952966369f;                         // 1/sqrt(32)
    float m = e;
    for (int off = 16; off; off >>= 1) m = fmaxf(m, __shfl_xor(m, off, 32));
    float ex = __expf(e - m);
    float sum = ex;
    for (int off = 16; off; off >>= 1) sum += __shfl_xor(sum, off, 32);
    attns[h][kk] = ex / sum;
  }
  __syncthreads();
  // o = attn @ v, residual
  {
    int h = t >> 5;
    float o = 0.f;
    for (int kk = 0; kk < K_; kk++) o += attns[h][kk] * vmat[kk][t];
    size_t off = (size_t)b * C_ * N_ + (size_t)t * N_ + i;
    r1[off] = x[off] + o;
  }
}

// ---------------- BatchNorm ----------------

__global__ void k_bnstats(const float* __restrict__ r, float* __restrict__ mean,
                          float* __restrict__ istd) {
  int c = blockIdx.x;
  int t = threadIdx.x, wave = t >> 5, lane = t & 31;
  float s = 0.f, s2 = 0.f;
  for (int b = 0; b < B_; b++) {
    const float* p = r + (size_t)b * C_ * N_ + (size_t)c * N_;
    for (int n = t; n < N_; n += 256) { float v = p[n]; s += v; s2 += v * v; }
  }
  for (int off = 16; off; off >>= 1) {
    s  += __shfl_xor(s, off, 32);
    s2 += __shfl_xor(s2, off, 32);
  }
  __shared__ float sh[8][2];
  if (lane == 0) { sh[wave][0] = s; sh[wave][1] = s2; }
  __syncthreads();
  if (t == 0) {
    float S = 0.f, S2 = 0.f;
    for (int w = 0; w < 8; w++) { S += sh[w][0]; S2 += sh[w][1]; }
    float m = S / (float)(B_ * N_);
    float var = S2 / (float)(B_ * N_) - m * m;
    mean[c] = m;
    istd[c] = rsqrtf(var + EPS_);
  }
}

// normalize + emit x1 [B,C,N] f32 (coalesced) and x1t_bf [B,N,C] bf16 via tiled transpose
__global__ void k_norm1(const float* __restrict__ r1, const float* __restrict__ g,
                        const float* __restrict__ bb, const float* __restrict__ mean,
                        const float* __restrict__ istd, float* __restrict__ x1,
                        bf16_t* __restrict__ x1t_bf) {
  __shared__ float tile[32][33];
  int n0 = blockIdx.x * 32, c0 = blockIdx.y * 32, b = blockIdx.z;
  int tx = threadIdx.x & 31, ty = threadIdx.x >> 5;
#pragma unroll
  for (int r = 0; r < 32; r += 8) {
    int c = c0 + ty + r;
    size_t o = ((size_t)(b * C_ + c)) * N_ + n0 + tx;
    float v = g[c] * (r1[o] - mean[c]) * istd[c] + bb[c];
    x1[o] = v;
    tile[ty + r][tx] = v;
  }
  __syncthreads();
#pragma unroll
  for (int r = 0; r < 32; r += 8)
    x1t_bf[((size_t)(b * N_ + n0 + ty + r)) * C_ + c0 + tx] = (bf16_t)tile[tx][ty + r];
}

__global__ void k_norm2(const float* __restrict__ r2, const float* __restrict__ g,
                        const float* __restrict__ bb, const float* __restrict__ mean,
                        const float* __restrict__ istd, float* __restrict__ out) {
  int tid = blockIdx.x * 256 + threadIdx.x;
  int c = (tid / N_) & (C_ - 1);
  out[tid] = g[c] * (r2[tid] - mean[c]) * istd[c] + bb[c];
}

// ---------------- fused MLP: w2 @ leaky(w1 @ x1) + x1 ----------------

__global__ void k_mlp(const bf16_t* __restrict__ x1t_bf, const bf16_t* __restrict__ w1_bf,
                      const bf16_t* __restrict__ w2_bf, const float* __restrict__ x1,
                      float* __restrict__ r2) {
  __shared__ bf16_t xstage[16][C_];                   // 4 KB point-major x1 tile
  __shared__ bf16_t hl[16][F_];                       // 16 KB hidden, point-major
  int t = threadIdx.x;
  int wave = t >> 5;
  int b = blockIdx.x >> 7;                            // N_/16 == 128 tiles per batch
  int n0 = (blockIdx.x & 127) * 16;
  const bf16_t* xrow = x1t_bf + ((size_t)(b * N_ + n0)) * C_;
  // async-stage the 16x128 bf16 x1t tile into LDS (ASYNCcnt-tracked)
  {
    unsigned ldsaddr = (unsigned)(uintptr_t)(&xstage[0][0]) + (unsigned)t * 16u;
    const bf16_t* gaddr = xrow + (t >> 4) * C_ + (t & 15) * 8;
    asm volatile("global_load_async_to_lds_b128 %0, %1, off"
                 :: "v"(ldsaddr), "v"(gaddr) : "memory");
    asm volatile("s_wait_asynccnt 0x0" ::: "memory");
  }
  __syncthreads();
  int mb, nl; cd_coords(mb, nl);
  // phase 1: h[o][n] = sum_c w1[o][c] * x1t[n][c]; LeakyReLU; 32 o-tiles over 8 waves
  for (int s = 0; s < 4; s++) {
    int o0 = (wave * 4 + s) * 16;
    v8f acc = {};
#pragma unroll
    for (int kt = 0; kt < 4; kt++) {
      v16bf a  = load_A16x32(w1_bf + (size_t)o0 * C_ + kt * 32, C_);
      v16bf bm = load_B32x16(&xstage[0][0] + kt * 32, C_);
      acc = WMMA_BF16(a, bm, acc);
    }
#pragma unroll
    for (int r = 0; r < 8; r++) {
      float v = acc[r];
      v = fmaxf(v, 0.2f * v);                         // LeakyReLU(0.2)
      hl[nl][o0 + mb + r] = (bf16_t)v;
    }
  }
  __syncthreads();
  // phase 2: f[c][n] = sum_o w2[c][o] * h[n][o]; one 16-row c-tile per wave
  {
    int c0 = wave * 16;
    v8f acc = {};
    for (int kt = 0; kt < 16; kt++) {
      v16bf a  = load_A16x32(w2_bf + (size_t)c0 * F_ + kt * 32, F_);
      v16bf bm = load_B32x16(&hl[0][0] + kt * 32, F_);
      acc = WMMA_BF16(a, bm, acc);
    }
#pragma unroll
    for (int r = 0; r < 8; r++) {
      int c = c0 + mb + r;
      size_t off = (size_t)b * C_ * N_ + (size_t)c * N_ + n0 + nl;
      r2[off] = x1[off] + acc[r];
    }
  }
}

// ---------------- host ----------------

extern "C" void kernel_launch(void* const* d_in, const int* in_sizes, int n_in,
                              void* d_out, int out_size, void* d_ws, size_t ws_size,
                              hipStream_t stream) {
  (void)in_sizes; (void)n_in; (void)out_size; (void)ws_size;
  const float* x  = (const float*)d_in[0];
  const float* wq = (const float*)d_in[1];
  const float* wk = (const float*)d_in[2];
  const float* wv = (const float*)d_in[3];
  const float* w1 = (const float*)d_in[4];
  const float* w2 = (const float*)d_in[5];
  const float* g1 = (const float*)d_in[6];
  const float* b1 = (const float*)d_in[7];
  const float* g2 = (const float*)d_in[8];
  const float* b2 = (const float*)d_in[9];

  char* ws = (char*)d_ws;
  size_t off = 0;
  auto alloc = [&](size_t bytes) -> char* {
    char* p = ws + off;
    off = (off + bytes + 255) & ~(size_t)255;
    return p;
  };
  float*    xt    = (float*)alloc((size_t)B_ * N_ * C_ * 4);
  bf16_t*   xtb   = (bf16_t*)alloc((size_t)B_ * N_ * C_ * 2);
  float*    x2    = (float*)alloc((size_t)B_ * N_ * 4);
  int*      idx   = (int*)alloc((size_t)B_ * N_ * K_ * 4);
  bf16_t*   wkb   = (bf16_t*)alloc((size_t)C_ * C_ * 2);
  bf16_t*   wvb   = (bf16_t*)alloc((size_t)C_ * C_ * 2);
  bf16_t*   w1b   = (bf16_t*)alloc((size_t)F_ * C_ * 2);
  bf16_t*   w2b   = (bf16_t*)alloc((size_t)C_ * F_ * 2);
  float*    m1    = (float*)alloc(C_ * 4);
  float*    istd1 = (float*)alloc(C_ * 4);
  float*    m2    = (float*)alloc(C_ * 4);
  float*    istd2 = (float*)alloc(C_ * 4);
  float*    r1    = (float*)alloc((size_t)B_ * C_ * N_ * 4);
  // big region: KNN key matrix, later overlaid by x1 / x1t_bf / r2 (keys dead by then)
  char*     big   = alloc((size_t)B_ * N_ * N_ * 4);
  unsigned* S     = (unsigned*)big;
  float*    x1    = (float*)big;
  bf16_t*   x1tb  = (bf16_t*)(big + (size_t)B_ * C_ * N_ * 4);
  float*    r2    = (float*)(big + (size_t)B_ * C_ * N_ * 4 + (size_t)B_ * N_ * C_ * 2);

  int total = B_ * C_ * N_;
  k_transpose<<<dim3(N_ / 32, C_ / 32, B_), 256, 0, stream>>>(x, xt, xtb);
  k_x2<<<B_ * N_ / 8, 256, 0, stream>>>(xt, x2);
  k_f2bf<<<(C_ * C_ + 255) / 256, 256, 0, stream>>>(wk, wkb, C_ * C_);
  k_f2bf<<<(C_ * C_ + 255) / 256, 256, 0, stream>>>(wv, wvb, C_ * C_);
  k_f2bf<<<(F_ * C_ + 255) / 256, 256, 0, stream>>>(w1, w1b, F_ * C_);
  k_f2bf<<<(C_ * F_ + 255) / 256, 256, 0, stream>>>(w2, w2b, C_ * F_);
  k_gram<<<dim3(N_ / 16, B_), 256, 0, stream>>>(xtb, x2, S);
  k_topk<<<B_ * N_ / 4, 128, 0, stream>>>(S, idx);
  k_attn<<<B_ * N_, 128, 0, stream>>>(x, xt, idx, wq, wkb, wvb, r1);
  k_bnstats<<<C_, 256, 0, stream>>>(r1, m1, istd1);
  k_norm1<<<dim3(N_ / 32, C_ / 32, B_), 256, 0, stream>>>(r1, g1, b1, m1, istd1, x1, x1tb);
  k_mlp<<<B_ * N_ / 16, 256, 0, stream>>>(x1tb, w1b, w2b, x1, r2);
  k_bnstats<<<C_, 256, 0, stream>>>(r2, m2, istd2);
  k_norm2<<<total / 256, 256, 0, stream>>>(r2, g2, b2, m2, istd2, (float*)d_out);
}
